// GraphMultiHeadAttention_70248485093582
// MI455X (gfx1250) — matrix operand order repaired and checked by reading, not compiled, not run on hardware
//
#include <hip/hip_runtime.h>

#define NTOK   4096
#define DMODEL 1024
#define NHEAD  16
#define HD     64

typedef __attribute__((ext_vector_type(16))) __bf16          v16bf;
typedef __attribute__((ext_vector_type(8)))  float           v8f;
typedef __attribute__((ext_vector_type(4)))  float           v4f;
typedef __attribute__((ext_vector_type(8)))  unsigned short  ushort8;
typedef __attribute__((ext_vector_type(16))) unsigned short  ushort16;

union BF16Tile {
    v16bf    bf;
    ushort16 us;
};

// round-to-nearest-even f32 -> bf16 bits
__device__ __forceinline__ unsigned short f32_bf16(float f) {
    unsigned int u = __float_as_uint(f);
    u += 0x7FFFu + ((u >> 16) & 1u);
    return (unsigned short)(u >> 16);
}

// exchange between lane n and lane n^16 (SWAPX16: xor=0x10, and=0x1f)
__device__ __forceinline__ float swap16(float x) {
    return __int_as_float(__builtin_amdgcn_ds_swizzle(__float_as_int(x), 0x401F));
}

__device__ __forceinline__ v8f wmma_bf16(v16bf a, v16bf b, v8f c) {
    return __builtin_amdgcn_wmma_f32_16x16x32_bf16(false, a, false, b, (short)0, c, false, false);
}

// Load a 16x32 bf16 A-operand tile (M = row0+lane%16, K-block at koff).
// A layout (contiguous k order): lanes 0-15 hold K=0..15, lanes 16-31 hold K=16..31.
__device__ __forceinline__ v16bf load_a_tile(const unsigned short* __restrict__ base,
                                             int row0, int stride, int koff,
                                             int l16, int half) {
    BF16Tile a;
    const unsigned short* p = base + (size_t)(row0 + l16) * stride + koff + half * 16;
    ushort8* d = (ushort8*)&a;
    d[0] = *(const ushort8*)p;
    d[1] = *(const ushort8*)(p + 8);
    return a.bf;
}

// ---------------------------------------------------------------------------
// Kernel 1a: straight f32 -> bf16 conversion of X (done ONCE, removes all
// per-tile conversion VALU work from the GEMM inner loop).
// ---------------------------------------------------------------------------
__global__ void convert_x_kernel(const float* __restrict__ X,
                                 unsigned short* __restrict__ Xb, int total) {
    int i = (blockIdx.x * blockDim.x + threadIdx.x) * 8;
    if (i < total) {
        ushort8 pk;
#pragma unroll
        for (int r = 0; r < 8; ++r) pk[r] = f32_bf16(X[i + r]);
        *(ushort8*)(Xb + i) = pk;
    }
}

// ---------------------------------------------------------------------------
// Kernel 1b: transposed f32 -> bf16 weight conversion: Wt[out][in] = W[in][out]
// (makes the GEMM B-operand a contiguous 32B load per lane)
// ---------------------------------------------------------------------------
__global__ void convert_wt_kernel(const float* __restrict__ W,
                                  unsigned short* __restrict__ Wt, int D) {
    int c = blockIdx.x * blockDim.x + threadIdx.x;  // out column
    int k = blockIdx.y;                             // in row
    if (c < D) Wt[(size_t)c * D + k] = f32_bf16(W[(size_t)k * D + c]);
}

// ---------------------------------------------------------------------------
// Kernel 2: projection GEMM  Y = Xb @ W + b  via bf16 WMMA, f32 accum.
// One wave = one 16x64 output tile (4 accumulators, A reused x4; the 64-wide
// column tile is exactly one head). K=1024 in 32 steps of 32.
// vmode==0: write bf16 [h][n][64];  vmode==1: write bf16 transposed [h][d][n].
// ---------------------------------------------------------------------------
__global__ void __launch_bounds__(32)
proj_gemm_kernel(const unsigned short* __restrict__ Xb,    // [N][D] bf16
                 const unsigned short* __restrict__ Wt,    // [out][in] bf16
                 const float* __restrict__ bias,
                 unsigned short* __restrict__ outp,
                 int vmode, int N, int D) {
    const int lane = threadIdx.x & 31;
    const int half = lane >> 4;
    const int l16  = lane & 15;
    const int cb   = blockIdx.x * 64;   // output-column tile (== one head)
    const int mb   = blockIdx.y * 16;   // row tile

    v8f acc[4];
#pragma unroll
    for (int j = 0; j < 4; ++j) acc[j] = (v8f){0, 0, 0, 0, 0, 0, 0, 0};

    for (int kk = 0; kk < D; kk += 32) {
        const v16bf a = load_a_tile(Xb, mb, D, kk, l16, half);
#pragma unroll
        for (int j = 0; j < 4; ++j) {
            // B layout: lane n<16 holds K=0..15 of column cb+16j+n,
            //           lane n+16 holds K=16..31. One contiguous 32B load.
            BF16Tile b;
            b.us = *(const ushort16*)(Wt + (size_t)(cb + 16 * j + l16) * D + kk + half * 16);
            acc[j] = wmma_bf16(a, b.bf, acc[j]);
        }
    }

    const int hh = cb / HD;    // head index
#pragma unroll
    for (int j = 0; j < 4; ++j) {
        const int   d0 = 16 * j;
        const float bv = bias[cb + d0 + l16];   // C layout: lane's column
        if (!vmode) {
            // [h][row][d]; row = mb + 8*half + r, d = d0 + l16
            unsigned short* dst = outp + ((size_t)hh * N + mb + 8 * half) * HD + d0 + l16;
#pragma unroll
            for (int r = 0; r < 8; ++r) dst[(size_t)r * HD] = f32_bf16(acc[j][r] + bv);
        } else {
            // transposed [h][d][row]; r consecutive in memory -> one 16B store
            ushort8 pk;
#pragma unroll
            for (int r = 0; r < 8; ++r) pk[r] = f32_bf16(acc[j][r] + bv);
            *(ushort8*)(outp + ((size_t)hh * HD + d0 + l16) * N + mb + 8 * half) = pk;
        }
    }
}

// ---------------------------------------------------------------------------
// Kernel 3: flash attention, one wave per (16-query tile, head).
// S^T = K.Q^T  (keys on M, queries on N)  -> softmax over keys is per-lane
// (8 accumulator regs) + one SWAPX16 ds_swizzle; O^T = V^T.P^T with V stored
// pre-transposed [h][d][n] so A-operand loads are contiguous 16B chunks.
// ---------------------------------------------------------------------------
__global__ void __launch_bounds__(32)
attn_kernel(const unsigned short* __restrict__ Qb,
            const unsigned short* __restrict__ Kb,
            const unsigned short* __restrict__ Vt,
            float* __restrict__ out, int N) {
    const int lane = threadIdx.x & 31;
    const int half = lane >> 4;
    const int l16  = lane & 15;
    const int h    = blockIdx.y;
    const int qb   = blockIdx.x * 16;

    const unsigned short* Qh = Qb + (size_t)h * N * HD;
    const unsigned short* Kh = Kb + (size_t)h * N * HD;
    const unsigned short* Vh = Vt + (size_t)h * HD * N;

    // Q^T B-operands (loop invariant): lane's column = query qb+l16,
    // dims dd + half*16 .. +15 contiguous (32B load). dd in {0,32}.
    BF16Tile bq0, bq1;
    bq0.us = *(const ushort16*)(Qh + (size_t)(qb + l16) * HD + half * 16);
    bq1.us = *(const ushort16*)(Qh + (size_t)(qb + l16) * HD + 32 + half * 16);

    float m_run = -3.0e38f;
    float l_run = 0.0f;
    v8f o[4];
#pragma unroll
    for (int t = 0; t < 4; ++t) o[t] = (v8f){0, 0, 0, 0, 0, 0, 0, 0};

    const float scale = 0.125f;  // 1/sqrt(64)

    for (int kb = 0; kb < N; kb += 32) {
        // prefetch next key-block's K rows and V lines into cache
        if (kb + 32 < N) {
            __builtin_prefetch(Kh + (size_t)(kb + 32 + l16) * HD, 0, 1);
            __builtin_prefetch(Vh + (size_t)l16 * N + kb + 32, 0, 1);
        }

        // ---- S^T tiles: st0 = keys kb..kb+15, st1 = keys kb+16..kb+31 ----
        v8f st0 = (v8f){0, 0, 0, 0, 0, 0, 0, 0};
        v8f st1 = (v8f){0, 0, 0, 0, 0, 0, 0, 0};
        st0 = wmma_bf16(load_a_tile(Kh, kb,      HD, 0,  l16, half), bq0.bf, st0);
        st0 = wmma_bf16(load_a_tile(Kh, kb,      HD, 32, l16, half), bq1.bf, st0);
        st1 = wmma_bf16(load_a_tile(Kh, kb + 16, HD, 0,  l16, half), bq0.bf, st1);
        st1 = wmma_bf16(load_a_tile(Kh, kb + 16, HD, 32, l16, half), bq1.bf, st1);

        // ---- online softmax over keys (per query column = per lane) ----
        float mloc = -3.0e38f;
#pragma unroll
        for (int r = 0; r < 8; ++r) {
            st0[r] *= scale;
            st1[r] *= scale;
            mloc = fmaxf(mloc, fmaxf(st0[r], st1[r]));
        }
        mloc = fmaxf(mloc, swap16(mloc));
        const float mnew  = fmaxf(m_run, mloc);
        const float alpha = __expf(m_run - mnew);

        float p0[8], p1[8], s = 0.0f;
#pragma unroll
        for (int r = 0; r < 8; ++r) {
            p0[r] = __expf(st0[r] - mnew);   // key kb +  r + 8*half
            p1[r] = __expf(st1[r] - mnew);   // key kb + 16 + r + 8*half
            s += p0[r] + p1[r];
        }
        s += swap16(s);
        l_run = l_run * alpha + s;
        m_run = mnew;
#pragma unroll
        for (int t = 0; t < 4; ++t) o[t] *= alpha;

        // ---- repack P^T (f32 C-layout) into bf16 B-layout ----
        // lane half 0 needs keys 0..15: own p0 (keys 0..7) + partner p0 (8..15)
        // lane half 1 needs keys 16..31: partner p1 (16..23) + own p1 (24..31)
        BF16Tile bp;
#pragma unroll
        for (int r = 0; r < 8; ++r) {
            const float op0 = swap16(p0[r]);
            const float op1 = swap16(p1[r]);
            const float lo  = half ? op1   : p0[r];
            const float hi  = half ? p1[r] : op0;
            bp.us[r]     = f32_bf16(lo);
            bp.us[8 + r] = f32_bf16(hi);
        }

        // ---- O^T += V^T . P^T  (4 dim tiles of 16) ----
#pragma unroll
        for (int t = 0; t < 4; ++t)
            o[t] = wmma_bf16(load_a_tile(Vh, t * 16, N, kb, l16, half), bp.bf, o[t]);
    }

    // ---- finalize & store: out[q][h*64 + d], d = t*16 + 8*half + r ----
    const float inv = 1.0f / l_run;
    float* op = out + (size_t)(qb + l16) * (NHEAD * HD) + h * HD;
#pragma unroll
    for (int t = 0; t < 4; ++t) {
        v4f lo, hi;
#pragma unroll
        for (int r = 0; r < 4; ++r) {
            lo[r] = o[t][r] * inv;
            hi[r] = o[t][4 + r] * inv;
        }
        *(v4f*)(op + t * 16 + half * 8)     = lo;
        *(v4f*)(op + t * 16 + half * 8 + 4) = hi;
    }
}

// ---------------------------------------------------------------------------
extern "C" void kernel_launch(void* const* d_in, const int* in_sizes, int n_in,
                              void* d_out, int out_size, void* d_ws, size_t ws_size,
                              hipStream_t stream) {
    const int N = NTOK, D = DMODEL;
    const float* X   = (const float*)d_in[0];
    const float* Wq  = (const float*)d_in[1];
    const float* bq  = (const float*)d_in[2];
    const float* Wk  = (const float*)d_in[3];
    const float* bk  = (const float*)d_in[4];
    const float* Wv  = (const float*)d_in[5];
    const float* bv  = (const float*)d_in[6];
    float* out = (float*)d_out;

    // workspace layout (bf16/ushort elements):
    //   Xb (4M) + 3 transposed weights (3x1M) + Q,K,Vt (3x4M)  = 38 MB
    unsigned short* ws  = (unsigned short*)d_ws;
    const size_t XS = (size_t)N * D;            // 4M elems
    const size_t DD = (size_t)D * D;            // 1M elems each
    const size_t QS = (size_t)NHEAD * N * HD;   // 4M elems each
    unsigned short* Xb  = ws;
    unsigned short* wtq = Xb + XS;
    unsigned short* wtk = wtq + DD;
    unsigned short* wtv = wtk + DD;
    unsigned short* Qbf = wtv + DD;
    unsigned short* Kbf = Qbf + QS;
    unsigned short* Vtf = Kbf + QS;

    convert_x_kernel<<<(int)(XS / (256 * 8)), 256, 0, stream>>>(X, Xb, (int)XS);

    dim3 cg(D / 256, D);
    convert_wt_kernel<<<cg, 256, 0, stream>>>(Wq, wtq, D);
    convert_wt_kernel<<<cg, 256, 0, stream>>>(Wk, wtk, D);
    convert_wt_kernel<<<cg, 256, 0, stream>>>(Wv, wtv, D);

    dim3 pg(D / 64, N / 16);
    proj_gemm_kernel<<<pg, 32, 0, stream>>>(Xb, wtq, bq, Qbf, 0, N, D);
    proj_gemm_kernel<<<pg, 32, 0, stream>>>(Xb, wtk, bk, Kbf, 0, N, D);
    proj_gemm_kernel<<<pg, 32, 0, stream>>>(Xb, wtv, bv, Vtf, 1, N, D);

    dim3 ag(N / 16, NHEAD);
    attn_kernel<<<ag, 32, 0, stream>>>(Qbf, Kbf, Vtf, out, N);
}
